// SparseMLPv2_66692252172319
// MI455X (gfx1250) — compile-verified, matrix operand order checked by reading, not run on hardware
//
#include <hip/hip_runtime.h>
#include <hip/hip_bf16.h>

// ---------------------------------------------------------------------------
// Grouped SwiGLU MLP for MI455X (gfx1250, wave32, WMMA + TDM).
//   T=16384 tokens, H=1024 hidden, E=8 experts, F=2048 ffn, T/E=2048.
// Pass 0: x/w1/w3 fp32 -> bf16; w2 fp32 -> bf16 TRANSPOSED to [E, H, F].
// Pass 1 (gate): gated = silu(x@w1^T) * (x@w3^T)  -> bf16 (d_ws)
// Pass 2 (down): out   = gated @ w2               -> f32  (d_out)
// All GEMM tile staging via Tensor Data Mover (tensor_load_to_lds, TENSORcnt)
// with double-buffered LDS so DMA overlaps WMMA compute. D# hardware padding
// produces the bank-conflict-free 72-elem LDS pitch.
// ---------------------------------------------------------------------------

#define HH    1024
#define EE    8
#define FF    2048
#define TPE   2048   // tokens per expert

#define BM 64
#define BN 64
#define BK 64
#define PB 72        // LDS row pitch (bf16 elems) = 144B: 16B-aligned, conflict-free

typedef __attribute__((ext_vector_type(16))) __bf16 v16bf;
typedef __attribute__((ext_vector_type(8)))  __bf16 v8bf;
typedef __attribute__((ext_vector_type(8)))  float  v8f;
typedef __attribute__((ext_vector_type(4)))  unsigned u32x4;
typedef __attribute__((ext_vector_type(8)))  int      i32x8;
typedef __attribute__((ext_vector_type(4)))  int      i32x4;

__device__ __forceinline__ unsigned short f2bf_bits(float f) {
    unsigned u = __builtin_bit_cast(unsigned, f);
    u += 0x7FFFu + ((u >> 16) & 1u);           // round-to-nearest-even
    return (unsigned short)(u >> 16);
}

// ---------------------------------------------------------------------------
// TDM: load a 64x64 bf16 tile (row stride = stride_elems) from global memory
// into LDS at lds_off, padding each 64-elem (32-DWORD) row by 4 DWORDs so the
// LDS pitch is 72 elems (144B). Wave-level; tracked by TENSORcnt.
// ---------------------------------------------------------------------------
__device__ __forceinline__ void tdm_load_tile64(unsigned lds_off,
                                                const __bf16* gsrc,
                                                unsigned stride_elems) {
    unsigned long long ga = (unsigned long long)gsrc;
    u32x4 g0;
    g0[0] = 1u;                                        // count=1, user mode
    g0[1] = lds_off;                                   // lds_addr
    g0[2] = (unsigned)(ga & 0xFFFFFFFFu);              // global_addr[31:0]
    g0[3] = (unsigned)((ga >> 32) & 0x01FFFFFFu)       // global_addr[56:32]
          | (2u << 30);                                // type = 2 ("image")
    i32x8 g1;
    g1[0] = (int)((1u << 16)                           // data_size = 2 bytes
                | (1u << 20)                           // pad_enable
                | (4u << 22)                           // pad_interval: 32 DWORDs
                | (3u << 25));                         // pad_amount: 4 DWORDs
    g1[1] = (int)(stride_elems << 16);                 // tensor_dim0[15:0]
    g1[2] = (int)(stride_elems >> 16)                  // tensor_dim0[31:16]
          | (int)(0xFFFFu << 16);                      // tensor_dim1[15:0] (large)
    g1[3] = (int)(64u << 16);                          // tensor_dim1 hi=0, tile_dim0=64
    g1[4] = 64;                                        // tile_dim1=64, tile_dim2=0
    g1[5] = (int)stride_elems;                         // tensor_dim0_stride[31:0]
    g1[6] = 0;                                         // stride hi, dim1_stride lo
    g1[7] = 0;
    i32x4 z = {0, 0, 0, 0};
#if defined(__clang_major__) && (__clang_major__ >= 23)
    i32x8 z8 = {0, 0, 0, 0, 0, 0, 0, 0};
    __builtin_amdgcn_tensor_load_to_lds(g0, g1, z, z, z8, 0);
#else
    __builtin_amdgcn_tensor_load_to_lds(g0, g1, z, z, 0);
#endif
}

__device__ __forceinline__ unsigned lds_offset(const void* p) {
    return (unsigned)(unsigned long long)p;            // low 32b = LDS offset
}

// Load a 16-elem bf16 fragment from LDS as two 16B halves.
__device__ __forceinline__ v16bf load_frag(const __bf16* p) {
    v8bf lo = *(const v8bf*)(p);
    v8bf hi = *(const v8bf*)(p + 8);
    v16bf r;
#pragma unroll
    for (int i = 0; i < 8; ++i) { r[i] = lo[i]; r[i + 8] = hi[i]; }
    return r;
}

// ---------------------------------------------------------------------------
// Pass 0a: fp32 -> bf16 bulk convert (float4 in, 8B out per thread).
// ---------------------------------------------------------------------------
__global__ __launch_bounds__(256) void cvt_kernel(const float* __restrict__ src,
                                                  __bf16* __restrict__ dst,
                                                  int n4) {
    int i = blockIdx.x * 256 + threadIdx.x;
    if (i < n4) {
        float4 v = ((const float4*)src)[i];
        ushort4 h;
        h.x = f2bf_bits(v.x); h.y = f2bf_bits(v.y);
        h.z = f2bf_bits(v.z); h.w = f2bf_bits(v.w);
        ((ushort4*)dst)[i] = h;
    }
}

// ---------------------------------------------------------------------------
// Pass 0b: w2 [E, F, H] fp32 -> [E, H, F] bf16 (LDS-tiled transpose, one pass).
// ---------------------------------------------------------------------------
__global__ __launch_bounds__(256) void cvt_transpose_kernel(
    const float* __restrict__ src,  // [E, F, H]
    __bf16* __restrict__ dst)       // [E, H, F]
{
    __shared__ alignas(16) __bf16 Ts[64 * PB];   // [h][f]

    const int tiles_h = HH / 64;                 // 16
    const int tiles_f = FF / 64;                 // 32
    int bid = blockIdx.x;
    int e   = bid / (tiles_f * tiles_h);
    int rr  = bid % (tiles_f * tiles_h);
    int f0  = (rr / tiles_h) * 64;
    int h0  = (rr % tiles_h) * 64;

    const float* se = src + ((size_t)e * FF + f0) * HH + h0;
    __bf16* de      = dst + ((size_t)e * HH + h0) * FF + f0;
    int tid = threadIdx.x;

    // Read 64x64 fp32 coalesced (float4), convert, scatter transposed into LDS.
#pragma unroll
    for (int i = 0; i < 4; ++i) {
        int linear = i * 256 + tid;              // 0..1023
        int fr_ = linear >> 4;                   // 16 float4 per f-row
        int h4  = (linear & 15) << 2;
        float4 v = *(const float4*)(se + (size_t)fr_ * HH + h4);
        Ts[(h4 + 0) * PB + fr_] = __builtin_bit_cast(__bf16, f2bf_bits(v.x));
        Ts[(h4 + 1) * PB + fr_] = __builtin_bit_cast(__bf16, f2bf_bits(v.y));
        Ts[(h4 + 2) * PB + fr_] = __builtin_bit_cast(__bf16, f2bf_bits(v.z));
        Ts[(h4 + 3) * PB + fr_] = __builtin_bit_cast(__bf16, f2bf_bits(v.w));
    }
    __syncthreads();

    // Write 64x64 bf16 coalesced (16B chunks) from LDS rows.
#pragma unroll
    for (int i = 0; i < 2; ++i) {
        int linear = i * 256 + tid;              // 0..511
        int hr = linear >> 3;                    // 8 chunks per h-row
        int c8 = (linear & 7) << 3;
        *(v8bf*)(de + (size_t)hr * FF + c8) = *(const v8bf*)(&Ts[hr * PB + c8]);
    }
}

// ---------------------------------------------------------------------------
// Pass 1: h1 = x@w1^T, h3 = x@w3^T, gated = silu(h1)*h3 (bf16).
// Double-buffered TDM pipeline: issue next tiles, wait for current (<=3),
// compute. All operands K(h)-contiguous.
// ---------------------------------------------------------------------------
__global__ __launch_bounds__(128) void gate_kernel(
    const __bf16* __restrict__ x,   // [E*TPE, H] bf16
    const __bf16* __restrict__ w1,  // [E, F, H] bf16
    const __bf16* __restrict__ w3,  // [E, F, H] bf16
    __bf16* __restrict__ gated)     // [E*TPE, F]
{
    __shared__ alignas(16) __bf16 Xs [2][BM * PB];
    __shared__ alignas(16) __bf16 W1s[2][BN * PB];
    __shared__ alignas(16) __bf16 W3s[2][BN * PB];

    const int tiles_n = FF / BN;                 // 32
    const int tiles_m = TPE / BM;                // 32
    int bid = blockIdx.x;
    int e   = bid / (tiles_m * tiles_n);
    int rr  = bid % (tiles_m * tiles_n);
    int bm  = (rr / tiles_n) * BM;
    int bn  = (rr % tiles_n) * BN;

    const __bf16* xe  = x  + ((size_t)e * TPE + bm) * HH;
    const __bf16* w1e = w1 + ((size_t)e * FF  + bn) * HH;
    const __bf16* w3e = w3 + ((size_t)e * FF  + bn) * HH;
    __bf16* og = gated + ((size_t)e * TPE + bm) * FF + bn;

    const int tid  = threadIdx.x;
    const int lane = tid & 31;
    const int wave = tid >> 5;                   // 4 waves: 2x2 over 64x64
    const int wm   = (wave >> 1) * 32;
    const int wn   = (wave & 1) * 32;
    const int fr   = lane & 15;
    const int fk   = (lane >> 4) << 4;           // 0 or 16

    unsigned xs_off[2]  = { lds_offset(&Xs [0][0]), lds_offset(&Xs [1][0]) };
    unsigned w1s_off[2] = { lds_offset(&W1s[0][0]), lds_offset(&W1s[1][0]) };
    unsigned w3s_off[2] = { lds_offset(&W3s[0][0]), lds_offset(&W3s[1][0]) };

    v8f acc1[2][2] = {};
    v8f acc3[2][2] = {};

    // Prologue: DMA first tiles into buffer 0.
    if (wave == 0) {
        tdm_load_tile64(xs_off[0],  xe,  HH);
        tdm_load_tile64(w1s_off[0], w1e, HH);
        tdm_load_tile64(w3s_off[0], w3e, HH);
    }

    int cur = 0;
    for (int k0 = 0; k0 < HH; k0 += BK) {
        if (wave == 0) {
            if (k0 + BK < HH) {                  // issue next, then wait current
                tdm_load_tile64(xs_off[cur ^ 1],  xe  + k0 + BK, HH);
                tdm_load_tile64(w1s_off[cur ^ 1], w1e + k0 + BK, HH);
                tdm_load_tile64(w3s_off[cur ^ 1], w3e + k0 + BK, HH);
                __builtin_amdgcn_s_wait_tensorcnt(3);
            } else {
                __builtin_amdgcn_s_wait_tensorcnt(0);
            }
        }
        __syncthreads();

        const __bf16* Xc  = &Xs [cur][0];
        const __bf16* W1c = &W1s[cur][0];
        const __bf16* W3c = &W3s[cur][0];
#pragma unroll
        for (int ks = 0; ks < 2; ++ks) {         // two 16x16x32 K substeps
            const int ko = ks * 32 + fk;
            v16bf a0 = load_frag(&Xc[(wm +  0 + fr) * PB + ko]);
            v16bf a1 = load_frag(&Xc[(wm + 16 + fr) * PB + ko]);
#pragma unroll
            for (int nt = 0; nt < 2; ++nt) {
                v16bf b1 = load_frag(&W1c[(wn + nt * 16 + fr) * PB + ko]);
                v16bf b3 = load_frag(&W3c[(wn + nt * 16 + fr) * PB + ko]);
                acc1[0][nt] = __builtin_amdgcn_wmma_f32_16x16x32_bf16(
                    false, a0, false, b1, (short)0, acc1[0][nt], false, false);
                acc3[0][nt] = __builtin_amdgcn_wmma_f32_16x16x32_bf16(
                    false, a0, false, b3, (short)0, acc3[0][nt], false, false);
                acc1[1][nt] = __builtin_amdgcn_wmma_f32_16x16x32_bf16(
                    false, a1, false, b1, (short)0, acc1[1][nt], false, false);
                acc3[1][nt] = __builtin_amdgcn_wmma_f32_16x16x32_bf16(
                    false, a1, false, b3, (short)0, acc3[1][nt], false, false);
            }
        }
        __syncthreads();
        cur ^= 1;
    }

    // SiLU gating epilogue; C layout: VGPR r -> M = r + 8*(lane>=16), N = lane%16.
    const int rofs = (lane >> 4) << 3;
#pragma unroll
    for (int mt = 0; mt < 2; ++mt)
#pragma unroll
        for (int nt = 0; nt < 2; ++nt)
#pragma unroll
            for (int r = 0; r < 8; ++r) {
                float h1v = acc1[mt][nt][r];
                float h3v = acc3[mt][nt][r];
                float g = (h1v / (1.0f + __expf(-h1v))) * h3v;
                int row = wm + mt * 16 + r + rofs;
                int col = wn + nt * 16 + (lane & 15);
                og[(size_t)row * FF + col] =
                    __builtin_bit_cast(__bf16, f2bf_bits(g));
            }
}

// ---------------------------------------------------------------------------
// Pass 2: out[t,h] = sum_f gated[t,f] * w2t[h,f].
// Both operands K(f)-contiguous (w2 was pre-transposed) -> both TDM-staged,
// double-buffered.
// ---------------------------------------------------------------------------
__global__ __launch_bounds__(128) void down_kernel(
    const __bf16* __restrict__ gated,  // [E*TPE, F] bf16
    const __bf16* __restrict__ w2t,    // [E, H, F] bf16 (transposed)
    float* __restrict__ out)           // [E*TPE, H] f32
{
    __shared__ alignas(16) __bf16 Gs [2][BM * PB];   // [t][f]
    __shared__ alignas(16) __bf16 W2s[2][BN * PB];   // [h][f]

    const int tiles_n = HH / BN;                 // 16
    const int tiles_m = TPE / BM;                // 32
    int bid = blockIdx.x;
    int e   = bid / (tiles_m * tiles_n);
    int rr  = bid % (tiles_m * tiles_n);
    int bm  = (rr / tiles_n) * BM;
    int bn  = (rr % tiles_n) * BN;

    const __bf16* ge   = gated + ((size_t)e * TPE + bm) * FF;
    const __bf16* w2te = w2t + ((size_t)e * HH + bn) * FF;
    float* oe = out + ((size_t)e * TPE + bm) * HH + bn;

    const int tid  = threadIdx.x;
    const int lane = tid & 31;
    const int wave = tid >> 5;
    const int wm   = (wave >> 1) * 32;
    const int wn   = (wave & 1) * 32;
    const int fr   = lane & 15;
    const int fk   = (lane >> 4) << 4;

    unsigned gs_off[2]  = { lds_offset(&Gs [0][0]), lds_offset(&Gs [1][0]) };
    unsigned w2s_off[2] = { lds_offset(&W2s[0][0]), lds_offset(&W2s[1][0]) };

    v8f acc[2][2] = {};

    if (wave == 0) {
        tdm_load_tile64(gs_off[0],  ge,   FF);
        tdm_load_tile64(w2s_off[0], w2te, FF);
    }

    int cur = 0;
    for (int k0 = 0; k0 < FF; k0 += BK) {        // 32 iterations
        if (wave == 0) {
            if (k0 + BK < FF) {
                tdm_load_tile64(gs_off[cur ^ 1],  ge   + k0 + BK, FF);
                tdm_load_tile64(w2s_off[cur ^ 1], w2te + k0 + BK, FF);
                __builtin_amdgcn_s_wait_tensorcnt(2);
            } else {
                __builtin_amdgcn_s_wait_tensorcnt(0);
            }
        }
        __syncthreads();

        const __bf16* Gc = &Gs [cur][0];
        const __bf16* Wc = &W2s[cur][0];
#pragma unroll
        for (int ks = 0; ks < 2; ++ks) {
            const int ko = ks * 32 + fk;
            v16bf a0 = load_frag(&Gc[(wm +  0 + fr) * PB + ko]);
            v16bf a1 = load_frag(&Gc[(wm + 16 + fr) * PB + ko]);
#pragma unroll
            for (int nt = 0; nt < 2; ++nt) {
                v16bf b = load_frag(&Wc[(wn + nt * 16 + fr) * PB + ko]);
                acc[0][nt] = __builtin_amdgcn_wmma_f32_16x16x32_bf16(
                    false, a0, false, b, (short)0, acc[0][nt], false, false);
                acc[1][nt] = __builtin_amdgcn_wmma_f32_16x16x32_bf16(
                    false, a1, false, b, (short)0, acc[1][nt], false, false);
            }
        }
        __syncthreads();
        cur ^= 1;
    }

    const int rofs = (lane >> 4) << 3;
#pragma unroll
    for (int mt = 0; mt < 2; ++mt)
#pragma unroll
        for (int nt = 0; nt < 2; ++nt)
#pragma unroll
            for (int r = 0; r < 8; ++r) {
                int row = wm + mt * 16 + r + rofs;
                int col = wn + nt * 16 + (lane & 15);
                oe[(size_t)row * HH + col] = acc[mt][nt][r];
            }
}

extern "C" void kernel_launch(void* const* d_in, const int* in_sizes, int n_in,
                              void* d_out, int out_size, void* d_ws, size_t ws_size,
                              hipStream_t stream) {
    const float* hs = (const float*)d_in[0];   // [T, H]
    const float* w1 = (const float*)d_in[1];   // [E, F, H]
    const float* w2 = (const float*)d_in[2];   // [E, F, H]
    const float* w3 = (const float*)d_in[3];   // [E, F, H]
    // d_in[4] expert_ids: tokens pre-sorted with equal per-expert load.
    (void)in_sizes; (void)n_in; (void)out_size; (void)ws_size;

    const size_t NTOK = (size_t)EE * TPE * HH;   // 16M elems (x)
    const size_t NW   = (size_t)EE * FF * HH;    // 16M elems (each weight)

    // Workspace layout (192 MB):
    //   [0,64M)    gated bf16 [T, F]
    //   [64M,96M)  x bf16, [96M,128M) w1 bf16, [128M,160M) w2t bf16 [E,H,F],
    //   [160M,192M) w3 bf16
    char* ws = (char*)d_ws;
    __bf16* gated = (__bf16*)(ws);
    __bf16* xb    = (__bf16*)(ws + ((size_t)64 << 20));
    __bf16* w1b   = (__bf16*)(ws + ((size_t)96 << 20));
    __bf16* w2tb  = (__bf16*)(ws + ((size_t)128 << 20));
    __bf16* w3b   = (__bf16*)(ws + ((size_t)160 << 20));
    float*  out   = (float*)d_out;

    dim3 cblk(256);
    dim3 cgrid((unsigned)((NTOK / 4 + 255) / 256));
    cvt_kernel<<<cgrid, cblk, 0, stream>>>(hs, xb, (int)(NTOK / 4));
    cvt_kernel<<<cgrid, cblk, 0, stream>>>(w1, w1b, (int)(NW / 4));
    cvt_kernel<<<cgrid, cblk, 0, stream>>>(w3, w3b, (int)(NW / 4));
    dim3 tgrid(EE * (FF / 64) * (HH / 64));      // 4096
    cvt_transpose_kernel<<<tgrid, cblk, 0, stream>>>(w2, w2tb);

    dim3 blk(128);
    dim3 grid_gate(EE * (TPE / BM) * (FF / BN));   // 8192
    dim3 grid_down(EE * (TPE / BM) * (HH / BN));   // 4096
    gate_kernel<<<grid_gate, blk, 0, stream>>>(xb, w1b, w3b, gated);
    down_kernel<<<grid_down, blk, 0, stream>>>(gated, w2tb, out);
}
